// InternLM2VEDecoderLayer_29420525977683
// MI455X (gfx1250) — matrix operand order repaired and checked by reading, not compiled, not run on hardware
//
#include <hip/hip_runtime.h>
#include <hip/hip_bf16.h>
#include <stdint.h>

// ---------------------------------------------------------------------------
// InternLM2-VE decoder layer for MI455X (gfx1250, wave32, WMMA).
// GEMMs: v_wmma_f32_16x16x32_bf16, 256x128 block tiles, double-buffered LDS.
//   A tile staged with global_load_async_to_lds_b128 (ASYNCcnt),
//   B tile (fp32 weights) register-prefetched + converted to bf16.
// Attention: flash-style causal GQA with register-prefetched K/V tiles.
// ---------------------------------------------------------------------------

#define S_LEN 2048
#define H_DIM 2048
#define NH    16
#define NKV   8
#define HD    128
#define FF_DIM 8192

typedef __bf16 bf16_t;
typedef __attribute__((ext_vector_type(16))) __bf16 v16bf;
typedef __attribute__((ext_vector_type(8)))  float  v8f;

union FragU { v16bf v; uint4 u[2]; };

__device__ __forceinline__ v8f zero8() {
  v8f z;
#pragma unroll
  for (int i = 0; i < 8; ++i) z[i] = 0.0f;
  return z;
}

#define WMMA_BF16(a, b, c) \
  __builtin_amdgcn_wmma_f32_16x16x32_bf16(false, (a), false, (b), (short)0, (c), false, false)

// Async DMA of 16 bytes/lane, global -> LDS (CDNA5 VGLOBAL opcode 98).
// LDS byte offset = low 32 bits of the generic pointer (ISA: LDS aperture
// places the offset in addr[31:0]).
__device__ __forceinline__ void async_ld_b128(void* lds, const void* g) {
  uint32_t l = (uint32_t)(uintptr_t)lds;
  uint64_t a = (uint64_t)(uintptr_t)g;
  asm volatile("global_load_async_to_lds_b128 %0, %1, off"
               :: "v"(l), "v"(a) : "memory");
}
#define WAIT_ASYNC0() asm volatile("s_wait_asynccnt 0x0" ::: "memory")

// A fragment (16x32, MxK), LDS row-major [m][k].
// ISA 7.12.2: lanes 0-15 -> M=lane, K 0-7 (v0-3) & 16-23 (v4-7);
//             lanes 16-31 -> K 8-15 / 24-31.
__device__ __forceinline__ v16bf load_frag_a(const bf16_t* lds, int mbase,
                                             int stride, int koff, int lane) {
  const int m  = lane & 15;
  const int kb = (lane >> 4) * 8;
  const bf16_t* p = lds + (size_t)(mbase + m) * stride + koff + kb;
  FragU f;
  f.u[0] = *(const uint4*)(p);
  f.u[1] = *(const uint4*)(p + 16);
  return f.v;
}

// B fragment (32x16, KxN), LDS stored N-major [n][k].
// lanes 0-15 -> N=lane, K 0-7 & 16-23; lanes 16-31 -> K 8-15 / 24-31.
__device__ __forceinline__ v16bf load_frag_b(const bf16_t* lds, int nbase,
                                             int stride, int koff, int lane) {
  const int n  = lane & 15;
  const int kh = (lane >> 4) * 8;
  const bf16_t* p = lds + (size_t)(nbase + n) * stride + koff + kh;
  FragU f;
  f.u[0] = *(const uint4*)(p);
  f.u[1] = *(const uint4*)(p + 16);
  return f.v;
}

// ---------------------------------------------------------------------------
// RMSNorm: one block per row; bf16 out (WMMA consumers) + optional fp32 copy.
// ---------------------------------------------------------------------------
__global__ __launch_bounds__(256) void rmsnorm_kernel(
    const float* __restrict__ x, const float* __restrict__ w,
    bf16_t* __restrict__ ob, float* __restrict__ of, int Hdim) {
  const int row = blockIdx.x;
  __shared__ float red[256];
  float s = 0.f;
  for (int c = threadIdx.x; c < Hdim; c += 256) {
    float v = x[(size_t)row * Hdim + c];
    s += v * v;
  }
  red[threadIdx.x] = s;
  __syncthreads();
  for (int st = 128; st > 0; st >>= 1) {
    if (threadIdx.x < st) red[threadIdx.x] += red[threadIdx.x + st];
    __syncthreads();
  }
  const float inv = rsqrtf(red[0] / (float)Hdim + 1e-6f);
  for (int c = threadIdx.x; c < Hdim; c += 256) {
    float v = x[(size_t)row * Hdim + c] * inv * w[c];
    ob[(size_t)row * Hdim + c] = (bf16_t)v;
    if (of) of[(size_t)row * Hdim + c] = v;
  }
}

// ---------------------------------------------------------------------------
// Tiled bf16 WMMA GEMM: C[M,N] = A[M,K](bf16) x B[K,N](fp32 -> bf16 in LDS).
// Block tile 256x128, BK=32, 512 threads = 16 waves (32x64 macro tile each).
// Double-buffered LDS; A via async-to-LDS DMA, B via register prefetch.
// ---------------------------------------------------------------------------
constexpr int EPI_BF16        = 0;  // store bf16
constexpr int EPI_ADDRES_F32  = 1;  // store f32 = acc + res
constexpr int EPI_SILU_MUL    = 2;  // store bf16 = silu(gate) * acc
constexpr int EPI_SCATTER_F32 = 3;  // store f32 at row rowmap[m]

template <int EPI>
__global__ __launch_bounds__(512) void gemm_bf16_kernel(
    const bf16_t* __restrict__ A, const float* __restrict__ Bw,
    int M, int N, int K,
    bf16_t* __restrict__ outb, float* __restrict__ outf,
    const float* __restrict__ res, const bf16_t* __restrict__ gate,
    const int* __restrict__ rowmap) {
  __shared__ alignas(16) bf16_t As[2][256 * 32];   // [m][k]
  __shared__ alignas(16) bf16_t Bs[2][128 * 32];   // [n][k]

  const int t    = threadIdx.x;
  const int lane = t & 31;
  const int wave = t >> 5;
  const int wm   = wave >> 1;        // 0..7  (M)
  const int wn   = wave & 1;         // 0..1  (N)
  const int bm   = blockIdx.y * 256;
  const int bn   = blockIdx.x * 128;

  v8f acc[2][4];
#pragma unroll
  for (int i = 0; i < 2; ++i)
#pragma unroll
    for (int j = 0; j < 4; ++j) acc[i][j] = zero8();

  // ---- staging helpers -------------------------------------------------
  // A: 256x32 bf16 = 1024 chunks of 16B; 2 per thread.
  auto issueA = [&](int kt, int buf) {
#pragma unroll
    for (int i = 0; i < 2; ++i) {
      int vi = i * 512 + t;
      int m  = vi >> 2;
      int kk = (vi & 3) * 8;
      async_ld_b128(&As[buf][m * 32 + kk],
                    A + (size_t)(bm + m) * K + kt + kk);
    }
  };
  // B: 128x32 fp32 = 4096 scalars; 8 per thread (register prefetch).
  float breg[8];
  auto fetchB = [&](int kt) {
#pragma unroll
    for (int i = 0; i < 8; ++i) {
      int e  = i * 512 + t;
      int n  = e & 127;
      int kk = e >> 7;
      breg[i] = Bw[(size_t)(kt + kk) * N + bn + n];
    }
  };
  auto storeB = [&](int buf) {
#pragma unroll
    for (int i = 0; i < 8; ++i) {
      int e  = i * 512 + t;
      int n  = e & 127;
      int kk = e >> 7;
      Bs[buf][n * 32 + kk] = (bf16_t)breg[i];
    }
  };

  // ---- prologue: stage tile 0 -----------------------------------------
  issueA(0, 0);
  fetchB(0);
  storeB(0);

  const int nkt = K >> 5;
  for (int it = 0; it < nkt; ++it) {
    const int cur = it & 1;
    const int nxt = cur ^ 1;
    WAIT_ASYNC0();        // A(cur) DMA complete (in-order ASYNCcnt)
    __syncthreads();      // all waves done with tile it-1; Bs(cur) visible

    const bool has_next = (it + 1) < nkt;
    if (has_next) {
      issueA((it + 1) * 32, nxt);  // DMA overlaps the WMMAs below
      fetchB((it + 1) * 32);       // vmem loads in flight during compute
    }

    v16bf afr[2], bfr[4];
#pragma unroll
    for (int i = 0; i < 2; ++i)
      afr[i] = load_frag_a(&As[cur][0], wm * 32 + i * 16, 32, 0, lane);
#pragma unroll
    for (int j = 0; j < 4; ++j)
      bfr[j] = load_frag_b(&Bs[cur][0], wn * 64 + j * 16, 32, 0, lane);
#pragma unroll
    for (int i = 0; i < 2; ++i)
#pragma unroll
      for (int j = 0; j < 4; ++j)
        acc[i][j] = WMMA_BF16(afr[i], bfr[j], acc[i][j]);

    if (has_next) storeB(nxt);  // disjoint buffer; read after next barrier
  }

  // Epilogue. C layout: VGPR r -> M = r + 8*(lane>=16), N = lane&15.
  const int halfsel = (lane >> 4) * 8;
  const int nin     = lane & 15;
#pragma unroll
  for (int i = 0; i < 2; ++i)
#pragma unroll
    for (int j = 0; j < 4; ++j) {
      const int row0 = bm + wm * 32 + i * 16;
      const int col0 = bn + wn * 64 + j * 16;
#pragma unroll
      for (int r = 0; r < 8; ++r) {
        const int grow = row0 + r + halfsel;
        const int gcol = col0 + nin;
        const float val = acc[i][j][r];
        if constexpr (EPI == EPI_BF16) {
          outb[(size_t)grow * N + gcol] = (bf16_t)val;
        } else if constexpr (EPI == EPI_ADDRES_F32) {
          outf[(size_t)grow * N + gcol] = val + res[(size_t)grow * N + gcol];
        } else if constexpr (EPI == EPI_SILU_MUL) {
          float g = (float)gate[(size_t)grow * N + gcol];
          float silu = g / (1.0f + __expf(-g));
          outb[(size_t)grow * N + gcol] = (bf16_t)(silu * val);
        } else {  // EPI_SCATTER_F32
          outf[(size_t)rowmap[grow] * N + gcol] = val;
        }
      }
    }
}

// ---------------------------------------------------------------------------
// RoPE (half rotation) applied in place to bf16 Q and K.
// ---------------------------------------------------------------------------
__global__ __launch_bounds__(64) void rope_kernel(
    bf16_t* __restrict__ q, bf16_t* __restrict__ k,
    const float* __restrict__ cosb, const float* __restrict__ sinb) {
  const int bid = blockIdx.x;
  const int s   = bid / (NH + NKV);
  const int hh  = bid % (NH + NKV);
  bf16_t* base = (hh < NH) ? (q + (size_t)s * (NH * HD) + hh * HD)
                           : (k + (size_t)s * (NKV * HD) + (hh - NH) * HD);
  const int d = threadIdx.x;  // 0..63
  const float x0 = (float)base[d];
  const float x1 = (float)base[d + 64];
  const float c0 = cosb[(size_t)s * HD + d];
  const float s0 = sinb[(size_t)s * HD + d];
  const float c1 = cosb[(size_t)s * HD + d + 64];
  const float s1 = sinb[(size_t)s * HD + d + 64];
  base[d]      = (bf16_t)(x0 * c0 - x1 * s0);
  base[d + 64] = (bf16_t)(x1 * c1 + x0 * s1);
}

// ---------------------------------------------------------------------------
// Flash-style causal GQA attention. Block = (64 q rows, head); 4 waves.
// K/V tiles register-prefetched one tile ahead to overlap fetch with WMMA.
// ---------------------------------------------------------------------------
__global__ __launch_bounds__(128) void flash_attn_kernel(
    const bf16_t* __restrict__ q, const bf16_t* __restrict__ k,
    const bf16_t* __restrict__ v, bf16_t* __restrict__ o) {
  __shared__ alignas(16) bf16_t Qs[64 * 128];   // [qrow][d]
  __shared__ alignas(16) bf16_t Ks[32 * 128];   // [kcol][d]
  __shared__ alignas(16) bf16_t Vt[128 * 32];   // [d][kcol] (transposed)
  __shared__ alignas(16) bf16_t Ps[4][16 * 32]; // per-wave P staging

  const int qt    = blockIdx.x;
  const int h     = blockIdx.y;
  const int t     = threadIdx.x;
  const int lane  = t & 31;
  const int wave  = t >> 5;
  const int qb    = qt * 64;
  const int qoff  = h * HD;
  const int kvoff = (h >> 1) * HD;   // GQA: kv head = h / 2

  // Q tile (64x128) -> LDS.
#pragma unroll
  for (int i = 0; i < 8; ++i) {
    int vi = i * 128 + t;
    int r  = vi >> 4;
    int dv = (vi & 15) * 8;
    *(uint4*)(Qs + r * 128 + dv) =
        *(const uint4*)(q + (size_t)(qb + r) * (NH * HD) + qoff + dv);
  }

  v8f o_acc[8];
  float mrow[8], lrow[8];
#pragma unroll
  for (int j = 0; j < 8; ++j) o_acc[j] = zero8();
#pragma unroll
  for (int r = 0; r < 8; ++r) { mrow[r] = -1e30f; lrow[r] = 0.f; }

  const float scale = 0.08838834764831845f;  // 1/sqrt(128)
  const int nkt = (qb + 64) >> 5;            // causal tile bound
  const int halfsel = (lane >> 4) * 8;
  const int ncol = lane & 15;

  // K/V register prefetch buffers.
  uint4 kreg[4];
  bf16_t vreg[32];
  auto fetchKV = [&](int kt) {
#pragma unroll
    for (int i = 0; i < 4; ++i) {
      int vi = i * 128 + t;
      int r  = vi >> 4;
      int dv = (vi & 15) * 8;
      kreg[i] = *(const uint4*)(k + (size_t)(kt * 32 + r) * (NKV * HD) + kvoff + dv);
    }
#pragma unroll
    for (int i = 0; i < 32; ++i) {
      int e  = i * 128 + t;
      int kc = e >> 7;
      int d  = e & 127;
      vreg[i] = v[(size_t)(kt * 32 + kc) * (NKV * HD) + kvoff + d];
    }
  };

  fetchKV(0);
  for (int kt = 0; kt < nkt; ++kt) {
    __syncthreads();   // LDS tile free (previous compute done)
#pragma unroll
    for (int i = 0; i < 4; ++i) {
      int vi = i * 128 + t;
      int r  = vi >> 4;
      int dv = (vi & 15) * 8;
      *(uint4*)(Ks + r * 128 + dv) = kreg[i];
    }
#pragma unroll
    for (int i = 0; i < 32; ++i) {
      int e  = i * 128 + t;
      int kc = e >> 7;
      int d  = e & 127;
      Vt[d * 32 + kc] = vreg[i];
    }
    __syncthreads();
    if (kt + 1 < nkt) fetchKV(kt + 1);   // overlap next fetch with compute

    // Scores: S[16q x 32k] = Q(16x128) . K^T.
    v8f s0 = zero8(), s1 = zero8();
#pragma unroll
    for (int dc = 0; dc < 4; ++dc) {
      v16bf aq = load_frag_a(Qs, wave * 16, 128, dc * 32, lane);
      v16bf b0 = load_frag_b(Ks, 0, 128, dc * 32, lane);
      v16bf b1 = load_frag_b(Ks, 16, 128, dc * 32, lane);
      s0 = WMMA_BF16(aq, b0, s0);
      s1 = WMMA_BF16(aq, b1, s1);
    }

    // Online softmax (row reductions across the 16-lane N groups).
#pragma unroll
    for (int r = 0; r < 8; ++r) {
      const int qrow = qb + wave * 16 + r + halfsel;
      float sc0 = s0[r] * scale;
      float sc1 = s1[r] * scale;
      if (kt * 32 + ncol > qrow)      sc0 = -1e30f;
      if (kt * 32 + 16 + ncol > qrow) sc1 = -1e30f;
      float tm = fmaxf(sc0, sc1);
#pragma unroll
      for (int m = 1; m < 16; m <<= 1) tm = fmaxf(tm, __shfl_xor(tm, m, 32));
      const float mnew = fmaxf(mrow[r], tm);
      const float corr = __expf(mrow[r] - mnew);
      const float p0 = __expf(sc0 - mnew);
      const float p1 = __expf(sc1 - mnew);
      float rs = p0 + p1;
#pragma unroll
      for (int m = 1; m < 16; m <<= 1) rs += __shfl_xor(rs, m, 32);
      lrow[r] = lrow[r] * corr + rs;
      mrow[r] = mnew;
#pragma unroll
      for (int j = 0; j < 8; ++j) o_acc[j][r] *= corr;
      const int pr = r + halfsel;
      Ps[wave][pr * 32 + ncol]      = (bf16_t)p0;
      Ps[wave][pr * 32 + 16 + ncol] = (bf16_t)p1;
    }
    asm volatile("s_wait_dscnt 0x0" ::: "memory");  // wave-private staging

    // O += P(16x32) . V(32x128)
    v16bf ap = load_frag_a(&Ps[wave][0], 0, 32, 0, lane);
#pragma unroll
    for (int j = 0; j < 8; ++j) {
      v16bf bv = load_frag_b(Vt, j * 16, 32, 0, lane);
      o_acc[j] = WMMA_BF16(ap, bv, o_acc[j]);
    }
  }

  // Normalize and store O (bf16) into [S, NH*HD].
#pragma unroll
  for (int r = 0; r < 8; ++r) {
    const float inv = 1.0f / lrow[r];
    const int row = qb + wave * 16 + r + halfsel;
#pragma unroll
    for (int j = 0; j < 8; ++j) {
      const int col = qoff + j * 16 + ncol;
      o[(size_t)row * (NH * HD) + col] = (bf16_t)(o_acc[j][r] * inv);
    }
  }
}

// ---------------------------------------------------------------------------
// Row gather: dst[r] = src[idx[r]] (bf16 rows of width Hdim).
// ---------------------------------------------------------------------------
__global__ __launch_bounds__(256) void gather_rows_kernel(
    const bf16_t* __restrict__ src, const int* __restrict__ idx,
    bf16_t* __restrict__ dst, int Hdim) {
  const int r = blockIdx.x;
  const int s = idx[r];
  for (int c = threadIdx.x; c < Hdim; c += 256)
    dst[(size_t)r * Hdim + c] = src[(size_t)s * Hdim + c];
}

// ---------------------------------------------------------------------------
// Host-side launch sequence.
// ---------------------------------------------------------------------------
extern "C" void kernel_launch(void* const* d_in, const int* in_sizes, int n_in,
                              void* d_out, int out_size, void* d_ws, size_t ws_size,
                              hipStream_t stream) {
  (void)in_sizes; (void)n_in; (void)out_size; (void)ws_size;

  const float* x        = (const float*)d_in[0];   // [1,S,H]
  const float* cosb     = (const float*)d_in[1];   // [S,HD]
  const float* sinb     = (const float*)d_in[2];   // [S,HD]
  const float* wq       = (const float*)d_in[3];
  const float* wk       = (const float*)d_in[4];
  const float* wv       = (const float*)d_in[5];
  const float* wo       = (const float*)d_in[6];
  const float* w1       = (const float*)d_in[7];
  const float* w3       = (const float*)d_in[8];
  const float* w2       = (const float*)d_in[9];
  const float* v1       = (const float*)d_in[10];
  const float* v3       = (const float*)d_in[11];
  const float* v2       = (const float*)d_in[12];
  const float* anw      = (const float*)d_in[13];
  const float* fnw      = (const float*)d_in[14];
  const int*   vis_idx  = (const int*)d_in[15];
  const int*   txt_idx  = (const int*)d_in[16];

  float* out   = (float*)d_out;
  float* resid = out + (size_t)S_LEN * H_DIM;

  size_t off = 0;
  auto carve = [&](size_t bytes) {
    void* p = (char*)d_ws + off;
    off += (bytes + 255) & ~(size_t)255;
    return p;
  };
  bf16_t* h1    = (bf16_t*)carve((size_t)S_LEN * H_DIM * 2);
  bf16_t* qbuf  = (bf16_t*)carve((size_t)S_LEN * NH  * HD * 2);
  bf16_t* kbuf  = (bf16_t*)carve((size_t)S_LEN * NKV * HD * 2);
  bf16_t* vbuf  = (bf16_t*)carve((size_t)S_LEN * NKV * HD * 2);
  bf16_t* attnb = (bf16_t*)carve((size_t)S_LEN * NH  * HD * 2);
  bf16_t* h2    = (bf16_t*)carve((size_t)S_LEN * H_DIM * 2);
  bf16_t* xg    = (bf16_t*)carve((size_t)1024 * H_DIM * 2);
  bf16_t* Gb    = (bf16_t*)carve((size_t)1024 * FF_DIM * 2);
  bf16_t* Pb    = (bf16_t*)carve((size_t)1024 * FF_DIM * 2);

  const dim3 blk512(512), blk256(256), blk128(128), blk64(64);

  // 1) attn RMSNorm -> h1 (bf16)
  rmsnorm_kernel<<<S_LEN, blk256, 0, stream>>>(x, anw, h1, nullptr, H_DIM);

  // 2) Q/K/V projections
  gemm_bf16_kernel<EPI_BF16><<<dim3((NH * HD) / 128, S_LEN / 256), blk512, 0, stream>>>(
      h1, wq, S_LEN, NH * HD, H_DIM, qbuf, nullptr, nullptr, nullptr, nullptr);
  gemm_bf16_kernel<EPI_BF16><<<dim3((NKV * HD) / 128, S_LEN / 256), blk512, 0, stream>>>(
      h1, wk, S_LEN, NKV * HD, H_DIM, kbuf, nullptr, nullptr, nullptr, nullptr);
  gemm_bf16_kernel<EPI_BF16><<<dim3((NKV * HD) / 128, S_LEN / 256), blk512, 0, stream>>>(
      h1, wv, S_LEN, NKV * HD, H_DIM, vbuf, nullptr, nullptr, nullptr, nullptr);

  // 3) RoPE in place on Q and K
  rope_kernel<<<S_LEN * (NH + NKV), blk64, 0, stream>>>(qbuf, kbuf, cosb, sinb);

  // 4) Causal GQA flash attention
  flash_attn_kernel<<<dim3(S_LEN / 64, NH), blk128, 0, stream>>>(qbuf, kbuf, vbuf, attnb);

  // 5) O projection + residual -> resid (fp32 output #2)
  gemm_bf16_kernel<EPI_ADDRES_F32><<<dim3(H_DIM / 128, S_LEN / 256), blk512, 0, stream>>>(
      attnb, wo, S_LEN, H_DIM, NH * HD, nullptr, resid, x, nullptr, nullptr);

  // 6) ffn RMSNorm -> h2 (bf16) and seed `out` with normed h (fp32)
  rmsnorm_kernel<<<S_LEN, blk256, 0, stream>>>(resid, fnw, h2, out, H_DIM);

  // 7) Vision MLP
  gather_rows_kernel<<<1024, blk256, 0, stream>>>(h2, vis_idx, xg, H_DIM);
  gemm_bf16_kernel<EPI_BF16><<<dim3(FF_DIM / 128, 1024 / 256), blk512, 0, stream>>>(
      xg, v1, 1024, FF_DIM, H_DIM, Gb, nullptr, nullptr, nullptr, nullptr);
  gemm_bf16_kernel<EPI_SILU_MUL><<<dim3(FF_DIM / 128, 1024 / 256), blk512, 0, stream>>>(
      xg, v3, 1024, FF_DIM, H_DIM, Pb, nullptr, nullptr, Gb, nullptr);
  gemm_bf16_kernel<EPI_SCATTER_F32><<<dim3(H_DIM / 128, 1024 / 256), blk512, 0, stream>>>(
      Pb, v2, 1024, H_DIM, FF_DIM, nullptr, out, nullptr, nullptr, vis_idx);

  // 8) Text MLP (reuses xg/Gb/Pb; stream-ordered)
  gather_rows_kernel<<<1024, blk256, 0, stream>>>(h2, txt_idx, xg, H_DIM);
  gemm_bf16_kernel<EPI_BF16><<<dim3(FF_DIM / 128, 1024 / 256), blk512, 0, stream>>>(
      xg, w1, 1024, FF_DIM, H_DIM, Gb, nullptr, nullptr, nullptr, nullptr);
  gemm_bf16_kernel<EPI_SILU_MUL><<<dim3(FF_DIM / 128, 1024 / 256), blk512, 0, stream>>>(
      xg, w3, 1024, FF_DIM, H_DIM, Pb, nullptr, nullptr, Gb, nullptr);
  gemm_bf16_kernel<EPI_SCATTER_F32><<<dim3(H_DIM / 128, 1024 / 256), blk512, 0, stream>>>(
      Pb, w2, 1024, H_DIM, FF_DIM, nullptr, out, nullptr, nullptr, txt_idx);
}
